// SPMLayer_79783312490994
// MI455X (gfx1250) — matrix operand order repaired and checked by reading, not compiled, not run on hardware
//
#include <hip/hip_runtime.h>
#include <hip/hip_bf16.h>
#include <math.h>

typedef __attribute__((ext_vector_type(16))) _Float16 v16h;
typedef __attribute__((ext_vector_type(8)))  _Float16 v8h;
typedef __attribute__((ext_vector_type(8)))  float    v8f;

#define BATCH   4
#define LSEQ    4096
#define CDIM    512
#define DMODEL  128
#define DINNER  256
#define DSTATE  16
#define DTRANK  8
#define SEQ     16          // 4 chunks * BATCH
#define DBC_N   40          // DT_RANK + 2*D_STATE

// ---------------------------------------------------------------------------
// f32 -> f16 weight conversion
// ---------------------------------------------------------------------------
__global__ void k_cvt_f16(const float* __restrict__ s, _Float16* __restrict__ d, int n) {
    int i = blockIdx.x * blockDim.x + threadIdx.x;
    if (i < n) d[i] = (_Float16)s[i];
}

// ---------------------------------------------------------------------------
// LayerNorm #1 over C=512 and split into 4 chunks of 128 (f16 for WMMA A)
// one block per (b,l); 128 threads x 4 channels
// ---------------------------------------------------------------------------
__global__ void k_ln1(const float* __restrict__ x, const float* __restrict__ g,
                      const float* __restrict__ be, _Float16* __restrict__ xn16) {
    __shared__ float red[128];
    int pos = blockIdx.x;          // b*LSEQ + l
    int b   = pos >> 12;
    int l   = pos & 4095;
    int tid = threadIdx.x;

    float v[4];
    float s = 0.f;
#pragma unroll
    for (int j = 0; j < 4; j++) {
        int c = tid + j * 128;
        v[j] = x[((size_t)(b * CDIM + c)) * LSEQ + l];
        s += v[j];
    }
    red[tid] = s; __syncthreads();
    for (int o = 64; o > 0; o >>= 1) { if (tid < o) red[tid] += red[tid + o]; __syncthreads(); }
    float mu = red[0] * (1.f / CDIM);
    __syncthreads();

    s = 0.f;
#pragma unroll
    for (int j = 0; j < 4; j++) { float dv = v[j] - mu; s += dv * dv; }
    red[tid] = s; __syncthreads();
    for (int o = 64; o > 0; o >>= 1) { if (tid < o) red[tid] += red[tid + o]; __syncthreads(); }
    float rinv = rsqrtf(red[0] * (1.f / CDIM) + 1e-5f);

#pragma unroll
    for (int j = 0; j < 4; j++) {
        int c = tid + j * 128;
        float y = (v[j] - mu) * rinv * g[c] + be[c];
        int q = c >> 7, d = c & 127;
        int sidx = q * BATCH + b;  // chunk-major sequence id
        xn16[((size_t)(sidx * LSEQ + l)) * DMODEL + d] = (_Float16)y;
    }
}

// ---------------------------------------------------------------------------
// WMMA GEMM (compile-time shapes):
//   C[M,N] = A[M,K] (f16, row-major) * W[N,K]^T (f16, row-major)
// Fragment layouts per CDNA5 ISA:
//   A 16x32 frag: lane m = l&15; K groups (l>>4)*8 .. +7 and +16 .. +23
//   B 32x16 frag: lane n = l&15; K groups (l>>4)*8 .. +7 and +16 .. +23 (same as A)
//   D:            VGPR r -> M = r + 8*(l>>4); N = l&15
// Each wave computes a 32x64 tile (2 M-tiles share the 4 B fragments).
// Non-multiple-of-16 N (x_proj, N=40): branch-free row clamp on B loads
// (padded columns are computed but never stored) -> no exec divergence.
// MODE 0: C[m*N+n] = acc
// MODE 1: (final proj) C[(b*N+n)*LSEQ + l] = acc + bias[n], m = b*LSEQ+l
// ---------------------------------------------------------------------------
__device__ __forceinline__ v16h wmma_load_a(const _Float16* __restrict__ A, int K,
                                            int m0, int k0, int lane) {
    int row = m0 + (lane & 15);
    int kb  = k0 + ((lane >> 4) << 3);
    const v8h* p0 = (const v8h*)(A + (size_t)row * K + kb);
    const v8h* p1 = (const v8h*)(A + (size_t)row * K + kb + 16);
    v8h lo = *p0, hi = *p1;
    return __builtin_shufflevector(lo, hi, 0,1,2,3,4,5,6,7,8,9,10,11,12,13,14,15);
}

template<int N, int K>
__device__ __forceinline__ v16h wmma_load_b(const _Float16* __restrict__ W,
                                            int n0, int k0, int lane) {
    int o = n0 + (lane & 15);
    if constexpr ((N & 15) != 0) o = (o < N) ? o : (N - 1);  // clamp; pads never stored
    int kk = k0 + ((lane >> 4) << 3);
    const v8h* p = (const v8h*)(W + (size_t)o * K + kk);
    v8h lo = p[0], hi = p[2];
    return __builtin_shufflevector(lo, hi, 0,1,2,3,4,5,6,7,8,9,10,11,12,13,14,15);
}

template<int N, int K, int MODE>
__global__ void k_wmma_gemm(const _Float16* __restrict__ A, const _Float16* __restrict__ W,
                            float* __restrict__ C, const float* __restrict__ bias) {
    constexpr int NT = (N >= 64) ? 4 : ((N + 15) / 16);   // n-subtiles per wave
    int lane = threadIdx.x & 31;
    int wave = threadIdx.x >> 5;
    int m0 = (blockIdx.x * 8 + wave * 2) * 16;   // 32 rows per wave, 128 per block
    int n0 = blockIdx.y * 64;

    v8f acc[2][NT];
#pragma unroll
    for (int i = 0; i < 2; i++)
#pragma unroll
        for (int t = 0; t < NT; t++)
#pragma unroll
            for (int r = 0; r < 8; r++) acc[i][t][r] = 0.f;

    for (int k0 = 0; k0 < K; k0 += 32) {
        if (k0 + 32 < K)
            __builtin_prefetch(A + (size_t)(m0 + (lane & 15)) * K + k0 + 32, 0, 3);
        v16h a0 = wmma_load_a(A, K, m0,      k0, lane);
        v16h a1 = wmma_load_a(A, K, m0 + 16, k0, lane);
#pragma unroll
        for (int t = 0; t < NT; t++) {
            v16h bf = wmma_load_b<N, K>(W, n0 + t * 16, k0, lane);
            acc[0][t] = __builtin_amdgcn_wmma_f32_16x16x32_f16(
                false, a0, false, bf, (short)0, acc[0][t], false, false);
            acc[1][t] = __builtin_amdgcn_wmma_f32_16x16x32_f16(
                false, a1, false, bf, (short)0, acc[1][t], false, false);
        }
    }

#pragma unroll
    for (int i = 0; i < 2; i++) {
        int mbase = m0 + i * 16 + ((lane >> 4) << 3);
#pragma unroll
        for (int t = 0; t < NT; t++) {
            int o = n0 + t * 16 + (lane & 15);
            if constexpr ((N & 15) != 0) { if (o >= N) continue; }
#pragma unroll
            for (int r = 0; r < 8; r++) {
                int m = mbase + r;
                float val = acc[i][t][r];
                if constexpr (MODE == 0) {
                    C[(size_t)m * N + o] = val;
                } else {
                    int bb = m >> 12;       // m = b*LSEQ + l
                    int l  = m & 4095;
                    C[((size_t)(bb * N + o)) * LSEQ + l] = val + bias[o];
                }
            }
        }
    }
}

// ---------------------------------------------------------------------------
// Causal depthwise conv (D_CONV=4, left pad 3) + bias + SiLU on xi = xz[:, :256]
// ---------------------------------------------------------------------------
__global__ void k_conv_silu(const float* __restrict__ xz, const float* __restrict__ cw,
                            const float* __restrict__ cb, float* __restrict__ xc,
                            _Float16* __restrict__ xc16) {
    size_t idx = (size_t)blockIdx.x * blockDim.x + threadIdx.x;   // SEQ*LSEQ*DINNER
    int e = (int)(idx & 255);
    size_t m = idx >> 8;           // s*LSEQ + l
    int l = (int)(m & 4095);
    float acc = cb[e];
#pragma unroll
    for (int j = 0; j < 4; j++) {
        int ll = l - 3 + j;
        if (ll >= 0) acc += cw[e * 4 + j] * xz[(m - (size_t)l + (size_t)ll) * 512 + e];
    }
    float sv = acc / (1.f + expf(-acc));   // SiLU
    xc[idx]   = sv;
    xc16[idx] = (_Float16)sv;
}

// ---------------------------------------------------------------------------
// dt projection (K = DT_RANK = 8) + softplus
// ---------------------------------------------------------------------------
__global__ void k_dtproj(const float* __restrict__ dbc, const float* __restrict__ dtw,
                         const float* __restrict__ dtb, float* __restrict__ dt) {
    size_t idx = (size_t)blockIdx.x * blockDim.x + threadIdx.x;
    int e = (int)(idx & 255);
    size_t m = idx >> 8;
    float acc = dtb[e];
#pragma unroll
    for (int r = 0; r < DTRANK; r++) acc += dbc[m * DBC_N + r] * dtw[e * DTRANK + r];
    dt[idx] = (acc > 20.f) ? acc : log1pf(expf(acc));
}

// ---------------------------------------------------------------------------
// Selective scan: block = one sequence, thread = one inner channel.
// h[16] lives in VGPRs; per-step shared B/C (32 floats) staged through LDS.
// ---------------------------------------------------------------------------
__global__ void k_scan(const float* __restrict__ dt, const float* __restrict__ dbc,
                       const float* __restrict__ xc, const float* __restrict__ A_log,
                       float* __restrict__ ys) {
    __shared__ float bc[32];
    int s = blockIdx.x;        // 0..15
    int d = threadIdx.x;       // 0..255
    float Av[DSTATE], h[DSTATE];
#pragma unroll
    for (int st = 0; st < DSTATE; st++) {
        Av[st] = -expf(A_log[d * DSTATE + st]);
        h[st]  = 0.f;
    }
    size_t base = (size_t)s * LSEQ;
    for (int l = 0; l < LSEQ; l++) {
        size_t m = base + l;
        if (d < 32) bc[d] = dbc[m * DBC_N + DTRANK + d];   // B (16) then C (16)
        __syncthreads();
        float dtv = dt[m * DINNER + d];
        float xv  = xc[m * DINNER + d];
        float dx  = dtv * xv;
        float y = 0.f;
#pragma unroll
        for (int st = 0; st < DSTATE; st++) {
            float dA = expf(dtv * Av[st]);
            h[st] = fmaf(dA, h[st], dx * bc[st]);
            y = fmaf(h[st], bc[16 + st], y);
        }
        ys[m * DINNER + d] = y;
        __syncthreads();
    }
}

// ---------------------------------------------------------------------------
// y = (ys + xc*D) * silu(z) -> f16 for out_proj A
// ---------------------------------------------------------------------------
__global__ void k_gate(const float* __restrict__ ys, const float* __restrict__ xc,
                       const float* __restrict__ Dp, const float* __restrict__ xz,
                       _Float16* __restrict__ y16) {
    size_t idx = (size_t)blockIdx.x * blockDim.x + threadIdx.x;
    int e = (int)(idx & 255);
    size_t m = idx >> 8;
    float z  = xz[m * 512 + 256 + e];
    float yv = (ys[idx] + xc[idx] * Dp[e]) * (z / (1.f + expf(-z)));
    y16[idx] = (_Float16)yv;
}

// ---------------------------------------------------------------------------
// skip add + chunk recombine + LayerNorm #2 -> f16 A for final projection
// ---------------------------------------------------------------------------
__global__ void k_combine_ln2(const float* __restrict__ ym, const _Float16* __restrict__ xn16,
                              const float* __restrict__ skip, const float* __restrict__ g,
                              const float* __restrict__ be, _Float16* __restrict__ y2n16) {
    __shared__ float red[128];
    int pos = blockIdx.x;      // b*LSEQ + l
    int b   = pos >> 12;
    int l   = pos & 4095;
    int tid = threadIdx.x;
    float sk = skip[0];

    float v[4];
    float s = 0.f;
#pragma unroll
    for (int j = 0; j < 4; j++) {
        int c = tid + j * 128;
        int q = c >> 7, d = c & 127;
        size_t mi = ((size_t)((q * BATCH + b) * LSEQ + l)) * DMODEL + d;
        v[j] = ym[mi] + sk * (float)xn16[mi];
        s += v[j];
    }
    red[tid] = s; __syncthreads();
    for (int o = 64; o > 0; o >>= 1) { if (tid < o) red[tid] += red[tid + o]; __syncthreads(); }
    float mu = red[0] * (1.f / CDIM);
    __syncthreads();

    s = 0.f;
#pragma unroll
    for (int j = 0; j < 4; j++) { float dv = v[j] - mu; s += dv * dv; }
    red[tid] = s; __syncthreads();
    for (int o = 64; o > 0; o >>= 1) { if (tid < o) red[tid] += red[tid + o]; __syncthreads(); }
    float rinv = rsqrtf(red[0] * (1.f / CDIM) + 1e-5f);

#pragma unroll
    for (int j = 0; j < 4; j++) {
        int c = tid + j * 128;
        float y = (v[j] - mu) * rinv * g[c] + be[c];
        y2n16[(size_t)pos * CDIM + c] = (_Float16)y;
    }
}

// ---------------------------------------------------------------------------
// launch
// ---------------------------------------------------------------------------
extern "C" void kernel_launch(void* const* d_in, const int* in_sizes, int n_in,
                              void* d_out, int out_size, void* d_ws, size_t ws_size,
                              hipStream_t stream) {
    const float* x         = (const float*)d_in[0];
    const float* norm_g    = (const float*)d_in[1];
    const float* norm_b    = (const float*)d_in[2];
    const float* in_proj_w = (const float*)d_in[3];
    const float* conv_w    = (const float*)d_in[4];
    const float* conv_b    = (const float*)d_in[5];
    const float* x_proj_w  = (const float*)d_in[6];
    const float* dt_proj_w = (const float*)d_in[7];
    const float* dt_proj_b = (const float*)d_in[8];
    const float* A_log     = (const float*)d_in[9];
    const float* D_param   = (const float*)d_in[10];
    const float* out_proj_w= (const float*)d_in[11];
    const float* proj_w    = (const float*)d_in[12];
    const float* proj_b    = (const float*)d_in[13];
    const float* skip_s    = (const float*)d_in[14];
    float* out = (float*)d_out;

    char* ws = (char*)d_ws;
    size_t off = 0;
    auto alloc = [&](size_t bytes) -> char* {
        char* p = ws + off;
        off += (bytes + 255) & ~(size_t)255;
        return p;
    };
    const size_t SL = (size_t)SEQ * LSEQ;     // 65536
    const size_t BL = (size_t)BATCH * LSEQ;   // 16384

    _Float16* w_in16  = (_Float16*)alloc((size_t)CDIM * DMODEL * 2);     // (512,128)
    _Float16* w_xp16  = (_Float16*)alloc((size_t)DBC_N * DINNER * 2);    // (40,256)
    _Float16* w_out16 = (_Float16*)alloc((size_t)DMODEL * DINNER * 2);   // (128,256)
    _Float16* w_pj16  = (_Float16*)alloc((size_t)CDIM * CDIM * 2);       // (512,512)
    _Float16* xn16    = (_Float16*)alloc(SL * DMODEL * 2);
    float*    xz      = (float*)   alloc(SL * 512 * 4);
    float*    xc      = (float*)   alloc(SL * DINNER * 4);
    _Float16* xc16    = (_Float16*)alloc(SL * DINNER * 2);
    float*    dbc     = (float*)   alloc(SL * DBC_N * 4);
    float*    dtv     = (float*)   alloc(SL * DINNER * 4);
    float*    ysv     = (float*)   alloc(SL * DINNER * 4);
    _Float16* y16     = (_Float16*)alloc(SL * DINNER * 2);
    float*    ym      = (float*)   alloc(SL * DMODEL * 4);
    _Float16* y2n16   = (_Float16*)alloc(BL * CDIM * 2);
    (void)ws_size; (void)n_in; (void)in_sizes; (void)out_size;

    // weight conversion to f16
    k_cvt_f16<<<(CDIM*DMODEL + 255)/256, 256, 0, stream>>>(in_proj_w,  w_in16,  CDIM*DMODEL);
    k_cvt_f16<<<(DBC_N*DINNER + 255)/256, 256, 0, stream>>>(x_proj_w,  w_xp16,  DBC_N*DINNER);
    k_cvt_f16<<<(DMODEL*DINNER + 255)/256, 256, 0, stream>>>(out_proj_w, w_out16, DMODEL*DINNER);
    k_cvt_f16<<<(CDIM*CDIM + 255)/256, 256, 0, stream>>>(proj_w,      w_pj16,  CDIM*CDIM);

    // layernorm 1 + chunk split
    k_ln1<<<(int)BL, 128, 0, stream>>>(x, norm_g, norm_b, xn16);

    // in_proj: (65536 x 128) x (512 x 128)^T -> xz
    k_wmma_gemm<512, 128, 0><<<dim3(512, 8), 128, 0, stream>>>(xn16, w_in16, xz, nullptr);

    // conv + silu
    k_conv_silu<<<(int)(SL * DINNER / 256), 256, 0, stream>>>(xz, conv_w, conv_b, xc, xc16);

    // x_proj: (65536 x 256) x (40 x 256)^T -> dbc
    k_wmma_gemm<DBC_N, 256, 0><<<dim3(512, 1), 128, 0, stream>>>(xc16, w_xp16, dbc, nullptr);

    // dt projection + softplus
    k_dtproj<<<(int)(SL * DINNER / 256), 256, 0, stream>>>(dbc, dt_proj_w, dt_proj_b, dtv);

    // selective scan
    k_scan<<<SEQ, DINNER, 0, stream>>>(dtv, dbc, xc, A_log, ysv);

    // gate: (ys + xc*D) * silu(z)
    k_gate<<<(int)(SL * DINNER / 256), 256, 0, stream>>>(ysv, xc, D_param, xz, y16);

    // out_proj: (65536 x 256) x (128 x 256)^T -> ym
    k_wmma_gemm<DMODEL, 256, 0><<<dim3(512, 2), 128, 0, stream>>>(y16, w_out16, ym, nullptr);

    // skip + recombine + layernorm 2
    k_combine_ln2<<<(int)BL, 128, 0, stream>>>(ym, xn16, skip_s, norm_g, norm_b, y2n16);

    // final projection with fused bias + NCHW transpose store
    k_wmma_gemm<CDIM, 512, 1><<<dim3(128, 8), 128, 0, stream>>>(y2n16, w_pj16, out, proj_b);
}
